// SCSA_13228499271840
// MI455X (gfx1250) — matrix operand order repaired and checked by reading, not compiled, not run on hardware
//
#include <hip/hip_runtime.h>
#include <hip/hip_bf16.h>
#include <math.h>
#include <stdint.h>

typedef __attribute__((ext_vector_type(16))) _Float16 v16h;
typedef __attribute__((ext_vector_type(8)))  float    v8f;
typedef __attribute__((ext_vector_type(4)))  float    v4f;

// Problem constants: x = (32, 512, 64, 64) fp32, G=8 -> 256 batch-groups of 64ch,
// split into two halves of CG=32 channels.
constexpr int NB    = 256;    // n*G
constexpr int CG    = 32;     // channels per half
constexpr int HDIM  = 64;
constexpr int WDIM  = 64;
constexpr int PLANE = HDIM * WDIM;   // 4096

__device__ __forceinline__ float sigmoidf_(float z) {
  return 1.0f / (1.0f + __expf(-z));
}

// ---------------------------------------------------------------------------
// Pass 1: streaming pooled statistics.
// One block per (b, ch) plane (16384 blocks).
// The 16KB plane is staged memory->LDS with CDNA5 async DMA loads
// (global_load_async_to_lds_b128, ASYNCcnt) -- no VGPR round trip.
//   ch <  32 (x0): gap[b][ch]  = mean over H,W
//   ch >= 32 (x1): rowm[b][c2][h] = mean over W
//                  colm[b][c2][w] = mean over H
//                  xs[b][c2]      = mean over H,W
// ---------------------------------------------------------------------------
__global__ void __launch_bounds__(256) scsa_stats_kernel(
    const float* __restrict__ x,
    float* __restrict__ rowm, float* __restrict__ colm,
    float* __restrict__ gap,  float* __restrict__ xs)
{
  const int plane = blockIdx.x;
  const int b  = plane >> 6;
  const int ch = plane & 63;
  const int t  = threadIdx.x;

  __shared__ float tile[PLANE];
  __shared__ float rowsum[HDIM];
  __shared__ float colsum[WDIM];

  // Async DMA the plane into LDS: 4 x (256 lanes x 16B) = 16KB.
  {
    const float* gbase = x + (size_t)plane * PLANE;        // SGPR 64-bit base
    const uint32_t lds0 = (uint32_t)(uintptr_t)(&tile[0]); // LDS byte offset
#pragma unroll
    for (int i = 0; i < 4; ++i) {
      uint32_t byteoff = (uint32_t)(t + 256 * i) * 16u;    // per-lane offset
      uint32_t ldsa    = lds0 + byteoff;
      asm volatile("global_load_async_to_lds_b128 %0, %1, %2 offset:0"
                   :: "v"(ldsa), "v"(byteoff), "s"(gbase) : "memory");
    }
    asm volatile("s_wait_asynccnt 0x0" ::: "memory");
  }
  __syncthreads();

  if (t < 64) {
    // row sums: swizzled column order -> conflict-free (bank = (t+i) mod 64)
    float s = 0.f;
#pragma unroll
    for (int i = 0; i < 64; ++i) {
      int w = (i + t) & 63;
      s += tile[t * 64 + w];
    }
    rowsum[t] = s;
  } else if (t < 128) {
    const int w = t - 64;
    float s = 0.f;
#pragma unroll
    for (int h = 0; h < 64; ++h) s += tile[h * 64 + w];
    colsum[w] = s;
  }
  __syncthreads();

  if (ch < CG) {
    if (t == 0) {
      float tot = 0.f;
      for (int i = 0; i < 64; ++i) tot += rowsum[i];
      gap[b * CG + ch] = tot * (1.0f / (float)PLANE);
    }
  } else {
    const int c2 = ch - CG;
    if (t < 64) {
      rowm[(b * CG + c2) * 64 + t] = rowsum[t] * (1.0f / 64.0f);
    } else if (t < 128) {
      colm[(b * CG + c2) * 64 + (t - 64)] = colsum[t - 64] * (1.0f / 64.0f);
    } else if (t == 128) {
      float tot = 0.f;
      for (int i = 0; i < 64; ++i) tot += rowsum[i];
      xs[b * CG + c2] = tot * (1.0f / (float)PLANE);
    }
  }
}

// ---------------------------------------------------------------------------
// Pass 2: per-batch WMMA pipeline, one 512-thread block (16 waves) per batch.
//   Y(32x128) = conv1_w(32x32) @ [rowm | colm](32x128) + b1   (16 wmma tiles)
//   GroupNorm per row over 128 positions, h-swish
//   a_h(32x64) = convh_w @ Y[:, :64] + bh                     (8 wmma tiles)
//   a_w(32x64) = convw_w @ Y[:, 64:] + bw                     (8 wmma tiles)
// All fragments gathered from LDS per the CDNA5 16-bit A/B VGPR layouts.
// ---------------------------------------------------------------------------
__global__ void __launch_bounds__(512) scsa_gemm_kernel(
    const float* __restrict__ rowm, const float* __restrict__ colm,
    const float* __restrict__ w1,  const float* __restrict__ b1,
    const float* __restrict__ gg,  const float* __restrict__ gb,
    const float* __restrict__ wh,  const float* __restrict__ bh,
    const float* __restrict__ wwp, const float* __restrict__ bw,
    float* __restrict__ ah_out, float* __restrict__ aw_out)
{
  __shared__ _Float16 sStat[CG][128];   // GEMM1 B-operand source (K x N)
  __shared__ _Float16 sW1[CG][CG];
  __shared__ _Float16 sWh[CG][CG];
  __shared__ _Float16 sWw[CG][CG];
  __shared__ float    sY[CG][128];      // GEMM1 f32 result
  __shared__ _Float16 sYh[CG][128];     // normalized+h-swish, f16 for GEMM2
  __shared__ float sB1[CG], sGg[CG], sGb[CG], sBh[CG], sBw[CG];
  __shared__ float sPart[CG][4], sPart2[CG][4];
  __shared__ float sScale[CG], sShift[CG];

  const int b    = blockIdx.x;
  const int t    = threadIdx.x;
  const int lane = t & 31;
  const int wave = t >> 5;
  const int lm   = lane & 15;   // M (A) / N (B,D) index within tile
  const int lhi  = lane >> 4;   // lane half select

  // ---- stage operands in LDS (f32 -> f16) ----
  const float* rb = rowm + b * CG * 64;
  const float* cbp = colm + b * CG * 64;
#pragma unroll
  for (int k = 0; k < 8; ++k) {
    int idx = t * 8 + k;              // 0..4095
    int i = idx >> 7;                 // input channel
    int p = idx & 127;                // pooled position
    float v = (p < 64) ? rb[i * 64 + p] : cbp[i * 64 + (p - 64)];
    sStat[i][p] = (_Float16)v;
  }
#pragma unroll
  for (int k = 0; k < 2; ++k) {
    int idx = t * 2 + k;              // 0..1023
    ((_Float16*)sW1)[idx] = (_Float16)w1[idx];
    ((_Float16*)sWh)[idx] = (_Float16)wh[idx];
    ((_Float16*)sWw)[idx] = (_Float16)wwp[idx];
  }
  if (t < CG) { sB1[t]=b1[t]; sGg[t]=gg[t]; sGb[t]=gb[t]; sBh[t]=bh[t]; sBw[t]=bw[t]; }
  __syncthreads();

  // ---- GEMM1: 2x8 tile grid, one tile per wave ----
  {
    const int mt = wave >> 3;         // 0..1
    const int nt = wave & 7;          // 0..7
    v16h a, bm;
#pragma unroll
    for (int j = 0; j < 16; ++j) {
      // A 16x32 f16 layout: lane<16 holds K {0..7,16..23}, lane>=16 holds K {8..15,24..31}
      int kA = (j & 7) + (lhi << 3) + ((j >> 3) << 4);
      a[j] = sW1[mt * 16 + lm][kA];
      // B 32x16 f16 layout: lane holds column lm; lane<16 holds K {0..7,16..23}, lane>=16 holds K {8..15,24..31}
      int kB = (j & 7) + (lhi << 3) + ((j >> 3) << 4);
      bm[j] = sStat[kB][nt * 16 + lm];
    }
    v8f c = {0.f,0.f,0.f,0.f,0.f,0.f,0.f,0.f};
    v8f d = __builtin_amdgcn_wmma_f32_16x16x32_f16(false, a, false, bm,
                                                   (short)0, c, false, false);
#pragma unroll
    for (int r = 0; r < 8; ++r) {
      // D f32 layout: VGPR r -> M = r + 8*lhi, N = lm
      int M = mt * 16 + (lhi << 3) + r;
      int N = nt * 16 + lm;
      sY[M][N] = d[r] + sB1[M];
    }
  }
  __syncthreads();

  // ---- GroupNorm stats: 4 partials per row ----
  if (t < 128) {
    const int row = t >> 2, q = t & 3;
    float s = 0.f, s2 = 0.f;
#pragma unroll
    for (int i = 0; i < 32; ++i) {
      float v = sY[row][q * 32 + i];
      s += v; s2 += v * v;
    }
    sPart[row][q] = s; sPart2[row][q] = s2;
  }
  __syncthreads();
  if (t < CG) {
    float s  = sPart[t][0] + sPart[t][1] + sPart[t][2] + sPart[t][3];
    float s2 = sPart2[t][0] + sPart2[t][1] + sPart2[t][2] + sPart2[t][3];
    float mu  = s * (1.0f / 128.0f);
    float var = s2 * (1.0f / 128.0f) - mu * mu;
    float inv = rsqrtf(var + 1e-5f);
    float sc  = sGg[t] * inv;
    sScale[t] = sc;
    sShift[t] = sGb[t] - mu * sc;
  }
  __syncthreads();

  // ---- normalize + h-swish -> f16 operand for GEMM2 ----
#pragma unroll
  for (int k = 0; k < 8; ++k) {
    int idx = t * 8 + k;
    int i = idx >> 7, p = idx & 127;
    float v  = sY[i][p] * sScale[i] + sShift[i];
    float cl = fminf(fmaxf(v + 3.0f, 0.0f), 6.0f);
    v = v * cl * (1.0f / 6.0f);
    sYh[i][p] = (_Float16)v;
  }
  __syncthreads();

  // ---- GEMM2: waves 0..7 -> a_h tiles, waves 8..15 -> a_w tiles (wave-uniform) ----
  {
    const bool isH = (wave < 8);
    const int  w2  = wave & 7;
    const int  mt  = w2 >> 2;        // 0..1
    const int  nt  = w2 & 3;         // 0..3
    const int  poff = isH ? 0 : 64;
    const _Float16 (*W)[CG] = isH ? (const _Float16 (*)[CG])sWh
                                  : (const _Float16 (*)[CG])sWw;
    v16h a, bm;
#pragma unroll
    for (int j = 0; j < 16; ++j) {
      int kA = (j & 7) + (lhi << 3) + ((j >> 3) << 4);
      a[j] = W[mt * 16 + lm][kA];
      int kB = (j & 7) + (lhi << 3) + ((j >> 3) << 4);
      bm[j] = sYh[kB][poff + nt * 16 + lm];
    }
    v8f c = {0.f,0.f,0.f,0.f,0.f,0.f,0.f,0.f};
    v8f d = __builtin_amdgcn_wmma_f32_16x16x32_f16(false, a, false, bm,
                                                   (short)0, c, false, false);
    float* outp = isH ? ah_out : aw_out;
#pragma unroll
    for (int r = 0; r < 8; ++r) {
      int M = mt * 16 + (lhi << 3) + r;
      int N = nt * 16 + lm;
      float bias = isH ? sBh[M] : sBw[M];
      outp[(b * CG + M) * 64 + N] = d[r] + bias;
    }
  }
}

// ---------------------------------------------------------------------------
// Pass 3: streaming apply + channel shuffle. One block per (b, ch) plane.
//   ch <  32: out = x0 * sigmoid(cweight*gap + cbias)
//   ch >= 32: out = x1 * sigmoid(a_h[h] * a_w[w] * x_s)
// x is 268MB and read once here (never again) -> non-temporal loads; out is
// write-once -> non-temporal stores (native ext-vector v4f for the builtin).
// Keeps the small reused stats buffers resident in L2.
// Destination plane index implements concat + final channel shuffle; planes
// stay contiguous so b128 accesses remain coalesced.
// ---------------------------------------------------------------------------
__global__ void __launch_bounds__(256) scsa_apply_kernel(
    const float* __restrict__ x,
    const float* __restrict__ cw, const float* __restrict__ cbv,
    const float* __restrict__ gap, const float* __restrict__ xs,
    const float* __restrict__ ah,  const float* __restrict__ aw,
    float* __restrict__ out)
{
  const int plane = blockIdx.x;
  const int b  = plane >> 6;
  const int ch = plane & 63;
  const int g  = b & 7;
  const int n  = b >> 3;
  const int cpre = g * 64 + ch;                 // channel before shuffle (0..511)
  const int cfin = ((cpre & 255) << 1) + (cpre >> 8);  // after shuffle
  const int t = threadIdx.x;

  const v4f* s4 = (const v4f*)(x + (size_t)plane * PLANE);
  v4f* d4 = (v4f*)(out + ((size_t)n * 512 + cfin) * PLANE);

  if (ch < CG) {
    const float fac = sigmoidf_(cw[ch] * gap[b * CG + ch] + cbv[ch]);
#pragma unroll
    for (int i = 0; i < 4; ++i) {
      v4f v = __builtin_nontemporal_load(&s4[t + 256 * i]);
      v *= fac;
      __builtin_nontemporal_store(v, &d4[t + 256 * i]);
    }
  } else {
    const int c2 = ch - CG;
    __shared__ float sah[HDIM];
    __shared__ float saw[WDIM];
    if (t < 64)       sah[t]      = ah[(b * CG + c2) * 64 + t];
    else if (t < 128) saw[t - 64] = aw[(b * CG + c2) * 64 + (t - 64)];
    const float s = xs[b * CG + c2];
    __syncthreads();
#pragma unroll
    for (int i = 0; i < 4; ++i) {
      int i4 = t + 256 * i;
      int e  = i4 << 2;          // element index: 4 consecutive w, same h
      int h  = e >> 6, w0 = e & 63;
      float ahs = sah[h] * s;
      v4f v = __builtin_nontemporal_load(&s4[i4]);
      v[0] *= sigmoidf_(ahs * saw[w0 + 0]);
      v[1] *= sigmoidf_(ahs * saw[w0 + 1]);
      v[2] *= sigmoidf_(ahs * saw[w0 + 2]);
      v[3] *= sigmoidf_(ahs * saw[w0 + 3]);
      __builtin_nontemporal_store(v, &d4[i4]);
    }
  }
}

// ---------------------------------------------------------------------------
extern "C" void kernel_launch(void* const* d_in, const int* in_sizes, int n_in,
                              void* d_out, int out_size, void* d_ws, size_t ws_size,
                              hipStream_t stream) {
  (void)in_sizes; (void)n_in; (void)out_size; (void)ws_size;
  const float* x   = (const float*)d_in[0];
  const float* cw  = (const float*)d_in[1];
  const float* cbv = (const float*)d_in[2];
  const float* w1  = (const float*)d_in[3];
  const float* b1  = (const float*)d_in[4];
  const float* gg  = (const float*)d_in[5];
  const float* gb  = (const float*)d_in[6];
  const float* wh  = (const float*)d_in[7];
  const float* bh  = (const float*)d_in[8];
  const float* wwp = (const float*)d_in[9];
  const float* bw  = (const float*)d_in[10];
  float* out = (float*)d_out;

  // workspace layout (floats): rowm, colm (2MB ea), gap, xs (32KB ea), a_h, a_w (2MB ea)
  float* ws   = (float*)d_ws;
  float* rowm = ws;
  float* colm = rowm + (size_t)NB * CG * 64;
  float* gap  = colm + (size_t)NB * CG * 64;
  float* xs   = gap  + (size_t)NB * CG;
  float* ah   = xs   + (size_t)NB * CG;
  float* aw   = ah   + (size_t)NB * CG * 64;

  scsa_stats_kernel<<<NB * 64, 256, 0, stream>>>(x, rowm, colm, gap, xs);
  scsa_gemm_kernel<<<NB, 512, 0, stream>>>(rowm, colm, w1, b1, gg, gb,
                                           wh, bh, wwp, bw, ah, aw);
  scsa_apply_kernel<<<NB * 64, 256, 0, stream>>>(x, cw, cbv, gap, xs, ah, aw, out);
}
